// SelfAttention_45792941310028
// MI455X (gfx1250) — matrix operand order repaired and checked by reading, not compiled, not run on hardware
//
#include <hip/hip_runtime.h>
#include <hip/hip_fp16.h>

typedef __attribute__((ext_vector_type(16))) _Float16 v16h;
typedef __attribute__((ext_vector_type(8)))  _Float16 v8h;
typedef __attribute__((ext_vector_type(8)))  float    v8f;
typedef __attribute__((ext_vector_type(2)))  __fp16   h2pk;   // return type of cvt_pkrtz

#define WMMA16(A,B,C) __builtin_amdgcn_wmma_f32_16x16x32_f16(false,(A),false,(B),(short)0,(C),false,false)
#define SHUF16(lo,hi) __builtin_shufflevector((lo),(hi),0,1,2,3,4,5,6,7,8,9,10,11,12,13,14,15)

// Problem constants
#define NB   4
#define NC   256
#define NN   4096          // 64*64
#define HP   66            // padded H/W
#define NCO  320           // 32(q)+32(k)+256(v) fused conv outputs
#define CT_N 20            // 320/16 out-channel tiles
// q pre-scaled by SCALE * log2(e) so softmax uses exp2 directly
__device__ __constant__ float QSCALE = 0.17677669529663687f * 1.4426950408889634f;

// -------- persistent scratch (fully rewritten every launch) --------
__device__ _Float16 g_xpad[NB * HP * HP * NC];          // NHWC f16, zero border
__device__ _Float16 g_wpack[CT_N * 9 * 8 * 512];        // B-frag layout: [frag][lane][16]
__device__ float    g_bias[NCO];
__device__ _Float16 g_q [NB * NN * 32];                 // [b][n][c] f16 (pre-scaled)
__device__ _Float16 g_k [NB * NN * 32];                 // [b][m][c] f16
__device__ _Float16 g_vn[(size_t)NB * NN * NC];         // [b][m][c] f16
__device__ _Float16 g_vc[(size_t)NB * NC * NN];         // [b][c][m] f16 (transposed)

// ===================== kernel 1: zero x_pad (borders) =====================
__global__ void k_zero_xpad(int n16) {
  int i = blockIdx.x * blockDim.x + threadIdx.x;
  if (i < n16) ((uint4*)g_xpad)[i] = make_uint4(0u, 0u, 0u, 0u);
}

// ===================== kernel 2: pack weights into B-fragment layout =====================
// frag = (ct*9 + tap)*8 + chunk ; per lane one contiguous v16h.
// B[k=ci][n=co]: lane<16 -> co=ct*16+lane, elem e -> ci=c0+e ; lane>=16 -> ci=c0+16+e
__global__ void k_pack_w(const float* __restrict__ Wq, const float* __restrict__ Wk,
                         const float* __restrict__ Wv) {
  int gid  = blockIdx.x * blockDim.x + threadIdx.x;
  int lane = gid & 31, frag = gid >> 5;
  if (frag >= CT_N * 9 * 8) return;
  int ct = frag / 72, j = (frag / 8) % 9, qc = frag & 7;
  int n = lane & 15, lh = lane >> 4;
  int co = ct * 16 + n, kh = j / 3, kw = j % 3;
  _Float16* dst = g_wpack + (size_t)frag * 512 + lane * 16;
  #pragma unroll
  for (int e = 0; e < 16; ++e) {
    int ci = qc * 32 + (lh ? 16 : 0) + e;
    float v;
    if (co < 32)      v = Wq[(((size_t)co       * 256 + ci) * 3 + kh) * 3 + kw];
    else if (co < 64) v = Wk[(((size_t)(co-32)  * 256 + ci) * 3 + kh) * 3 + kw];
    else              v = Wv[(((size_t)(co-64)  * 256 + ci) * 3 + kh) * 3 + kw];
    dst[e] = (_Float16)v;
  }
}

__global__ void k_pack_b(const float* bq, const float* bk, const float* bv) {
  int i = blockIdx.x * blockDim.x + threadIdx.x;
  if (i < 32)       g_bias[i] = bq[i];
  else if (i < 64)  g_bias[i] = bk[i - 32];
  else if (i < 320) g_bias[i] = bv[i - 64];
}

// ===================== kernel 3: NCHW f32 -> padded NHWC f16 =====================
__global__ __launch_bounds__(256) void k_pad_x(const float* __restrict__ x) {
  __shared__ float tile[64][65];
  int bx = blockIdx.x;                       // NB * 64(ptile) * 4(ctile)
  int ctile = bx & 3, ptile = (bx >> 2) & 63, b = bx >> 8;
  int c0 = ctile * 64, p0 = ptile * 64;
  int tx = threadIdx.x & 63, ty = threadIdx.x >> 6;
  const float* xs = x + (size_t)b * NC * NN;
  #pragma unroll
  for (int i = 0; i < 16; ++i) {
    int c = ty + 4 * i;
    tile[c][tx] = xs[(size_t)(c0 + c) * NN + p0 + tx];   // 256B coalesced reads
  }
  __syncthreads();
  #pragma unroll
  for (int i = 0; i < 16; ++i) {
    int pr = ty + 4 * i, p = p0 + pr, h = p >> 6, w = p & 63;
    g_xpad[((size_t)(b * HP + h + 1) * HP + (w + 1)) * NC + c0 + tx] = (_Float16)tile[tx][pr];
  }
}

// ===================== kernel 4: fused implicit-GEMM conv (q|k|v) =====================
// 1 wave: 16 pixels (one W-run) x 16 out-channels; K = 9 taps * 256 ci = 72 WMMAs.
__global__ __launch_bounds__(32) void k_conv() {
  int bx = blockIdx.x;
  int ct = bx % CT_N, pix = bx / CT_N;
  int wt = pix & 3, h = (pix >> 2) & 63, b = pix >> 8;
  int lane = threadIdx.x, ln = lane & 15, lh = lane >> 4;
  v8f acc = {};
  #pragma unroll
  for (int j = 0; j < 9; ++j) {
    int gh = h + j / 3, gw = wt * 16 + ln + j % 3;
    const _Float16* base = g_xpad + ((size_t)(b * HP + gh) * HP + gw) * NC;
    const _Float16* wb   = g_wpack + (size_t)((ct * 9 + j) * 8) * 512 + lane * 16;
    #pragma unroll
    for (int qc = 0; qc < 8; ++qc) {
      int ao = qc * 32 + (lh ? 16 : 0);                // A layout: lane<16 K{0-15}, else K{16-31}
      v8h lo = *(const v8h*)(base + ao);
      v8h hi = *(const v8h*)(base + ao + 8);
      v16h a  = SHUF16(lo, hi);
      v16h bf = *(const v16h*)(wb + qc * 512);
      acc = WMMA16(a, bf, acc);
    }
  }
  float bias = g_bias[ct * 16 + ln];
  int n0 = h * 64 + wt * 16;
  #pragma unroll
  for (int r = 0; r < 8; ++r) {                        // D: row = r + 8*lh, col = ln
    int n = n0 + r + 8 * lh;
    float val = acc[r] + bias;
    if (ct < 2)      g_q [((size_t)(b * NN + n)) * 32 + ct * 16 + ln]        = (_Float16)(val * QSCALE);
    else if (ct < 4) g_k [((size_t)(b * NN + n)) * 32 + (ct - 2) * 16 + ln]  = (_Float16)val;
    else             g_vn[((size_t)(b * NN + n)) * NC + (ct - 4) * 16 + ln]  = (_Float16)val;
  }
}

// ===================== kernel 5: v [b][m][c] -> [b][c][m] =====================
__global__ __launch_bounds__(256) void k_trans_v() {
  __shared__ _Float16 tile[64][65];
  int bx = blockIdx.x;                       // NB * 64(mtile) * 4(ctile)
  int ctile = bx & 3, mtile = (bx >> 2) & 63, b = bx >> 8;
  int c0 = ctile * 64, m0 = mtile * 64;
  int tx = threadIdx.x & 63, ty = threadIdx.x >> 6;
  #pragma unroll
  for (int i = 0; i < 16; ++i) {
    int m = ty + 4 * i;
    tile[m][tx] = g_vn[((size_t)(b * NN) + m0 + m) * NC + c0 + tx];
  }
  __syncthreads();
  #pragma unroll
  for (int i = 0; i < 16; ++i) {
    int c = ty + 4 * i;
    g_vc[((size_t)(b * NC) + c0 + c) * NN + m0 + tx] = tile[tx][c];
  }
}

// ===================== kernel 6: fused flash attention + residual =====================
// 1 wave: 16 queries x 128 channels. Per 32-key step: 2 QK WMMAs + 8 PV WMMAs.
// K tiles use even/odd column interleave so exp(S) packs (pk_f16) into b32 LDS
// writes; P comes back as the A-fragment via two ds_read_b128 per lane.
__global__ __launch_bounds__(128) void k_attn(const float* __restrict__ x,
                                              const float* __restrict__ gptr,
                                              float* __restrict__ out) {
  __shared__ _Float16 Plds[4][512];        // per-wave 16x32 f16 P tile
  __shared__ float    Ost[4][128 * 20];    // per-wave 128c x 16n staging (pad 20)
  int wave = threadIdx.x >> 5, lane = threadIdx.x & 31;
  int ln = lane & 15, lh = lane >> 4;
  int g = blockIdx.x * 4 + wave;
  int b = g >> 9, rem = g & 511, qt = rem >> 1, chH = rem & 1;
  int n0 = qt * 16, cb = chH * 128;

  // Q A-fragment (row = ln for both halves; K split per half)
  const _Float16* qp = g_q + ((size_t)(b * NN + n0) + ln) * 32 + (lh ? 16 : 0);
  v16h aq = SHUF16(*(const v8h*)qp, *(const v8h*)(qp + 8));

  const _Float16* kbase = g_k + (size_t)b * NN * 32;
  const _Float16* vbase = g_vc + ((size_t)(b * NC + cb)) * NN;

  v8f O[8];
  float rmax[8], lsum[8];
  #pragma unroll
  for (int r = 0; r < 8; ++r) { O[r] = (v8f){}; rmax[r] = -1e30f; lsum[r] = 0.f; }

  _Float16* P  = Plds[wave];
  unsigned* P32 = (unsigned*)P;

  for (int it = 0; it < 128; ++it) {
    int m0 = it * 32;
    // K B-frags: col j of tile0 -> key m0+2j, tile1 -> m0+2j+1 (contiguous v16h per lane)
    const _Float16* kp = kbase + (size_t)(m0 + 2 * ln) * 32 + lh * 16;
    v16h bk0 = *(const v16h*)kp;
    v16h bk1 = *(const v16h*)(kp + 32);
    v8f z = {};
    v8f s0 = WMMA16(aq, bk0, z);
    v8f s1 = WMMA16(aq, bk1, z);

    // online row-max (rows 0-7 in lanes 0-15, rows 8-15 in lanes 16-31)
    float nm[8]; bool chg = false;
    #pragma unroll
    for (int r = 0; r < 8; ++r) {
      float t = fmaxf(s0[r], s1[r]);
      t = fmaxf(t, __shfl_xor(t, 1, 32));
      t = fmaxf(t, __shfl_xor(t, 2, 32));
      t = fmaxf(t, __shfl_xor(t, 4, 32));
      t = fmaxf(t, __shfl_xor(t, 8, 32));
      nm[r] = fmaxf(rmax[r], t);
      chg = chg || (nm[r] > rmax[r]);
    }
    if (__ballot(chg)) {                   // skip rescale when max unchanged (common case)
      #pragma unroll
      for (int r = 0; r < 8; ++r) {
        float al = __builtin_amdgcn_exp2f(rmax[r] - nm[r]);
        rmax[r] = nm[r];
        lsum[r] *= al;
        #pragma unroll
        for (int ctl = 0; ctl < 8; ++ctl) O[ctl][r] *= al;
      }
    }
    // P = exp2(S - max), packed pairs (even,odd col) -> b32 LDS, natural [n][m0..31] row
    #pragma unroll
    for (int r = 0; r < 8; ++r) {
      float p0 = __builtin_amdgcn_exp2f(s0[r] - rmax[r]);
      float p1 = __builtin_amdgcn_exp2f(s1[r] - rmax[r]);
      lsum[r] += p0 + p1;
      union { h2pk h; unsigned u; } cv;
      cv.h = __builtin_amdgcn_cvt_pkrtz(p0, p1);
      P32[(r + 8 * lh) * 16 + ln] = cv.u;
    }
    // P back as A-fragment (same-wave LDS, in order)
    v16h ap = SHUF16(*(const v8h*)(P + ln * 32 + lh * 16),
                     *(const v8h*)(P + ln * 32 + lh * 16 + 8));
    #pragma unroll
    for (int ctl = 0; ctl < 8; ++ctl) {
      v16h bv = *(const v16h*)(vbase + (size_t)(ctl * 16 + ln) * NN + m0 + lh * 16);
      O[ctl] = WMMA16(ap, bv, O[ctl]);
    }
  }

  // finalize: l = sum over row, scale by gamma / l
  float gm = gptr[0];
  float linv[8];
  #pragma unroll
  for (int r = 0; r < 8; ++r) {
    float l = lsum[r];
    l += __shfl_xor(l, 1, 32);
    l += __shfl_xor(l, 2, 32);
    l += __shfl_xor(l, 4, 32);
    l += __shfl_xor(l, 8, 32);
    linv[r] = gm * __builtin_amdgcn_rcpf(l);
  }
  float* os = Ost[wave];
  #pragma unroll
  for (int ctl = 0; ctl < 8; ++ctl)
    #pragma unroll
    for (int r = 0; r < 8; ++r)
      os[(ctl * 16 + ln) * 20 + r + 8 * lh] = O[ctl][r] * linv[r];

  // residual + coalesced NCHW store (same-wave LDS, 64B chunks)
  size_t obase = ((size_t)b * NC + cb) * NN + n0;
  #pragma unroll
  for (int pass = 0; pass < 16; ++pass) {
    int row = pass * 8 + (lane >> 2);
    int qn  = (lane & 3) * 4;
    size_t idx = obase + (size_t)row * NN + qn;
    float4 xv = *(const float4*)(x + idx);
    float4 o;
    o.x = os[row * 20 + qn + 0] + xv.x;
    o.y = os[row * 20 + qn + 1] + xv.y;
    o.z = os[row * 20 + qn + 2] + xv.z;
    o.w = os[row * 20 + qn + 3] + xv.w;
    *(float4*)(out + idx) = o;
  }
}

// ===================== host =====================
extern "C" void kernel_launch(void* const* d_in, const int* in_sizes, int n_in,
                              void* d_out, int out_size, void* d_ws, size_t ws_size,
                              hipStream_t stream) {
  (void)in_sizes; (void)n_in; (void)d_ws; (void)ws_size; (void)out_size;
  const float* x  = (const float*)d_in[0];
  const float* Wq = (const float*)d_in[1];
  const float* bq = (const float*)d_in[2];
  const float* Wk = (const float*)d_in[3];
  const float* bk = (const float*)d_in[4];
  const float* Wv = (const float*)d_in[5];
  const float* bv = (const float*)d_in[6];
  const float* gm = (const float*)d_in[7];
  float* out = (float*)d_out;

  int n16 = (NB * HP * HP * NC * 2) / 16;                 // x_pad bytes / 16
  k_zero_xpad<<<(n16 + 255) / 256, 256, 0, stream>>>(n16);
  k_pack_w<<<(CT_N * 9 * 8 * 32 + 255) / 256, 256, 0, stream>>>(Wq, Wk, Wv);
  k_pack_b<<<2, 256, 0, stream>>>(bq, bk, bv);
  k_pad_x<<<NB * 64 * 4, 256, 0, stream>>>(x);
  k_conv<<<NB * 256 * CT_N, 32, 0, stream>>>();
  k_trans_v<<<NB * 64 * 4, 256, 0, stream>>>();
  k_attn<<<(NB * 256 * 2) / 4, 128, 0, stream>>>(x, gm, out);
}
